// ImprovedGATRegressor_67534065762831
// MI455X (gfx1250) — matrix operand order repaired and checked by reading, not compiled, not run on hardware
//
#include <hip/hip_runtime.h>
#include <hip/hip_bf16.h>

#define N_NODES   10000
#define N_EDGES   100000
#define IN_CH     64
#define EDGE_DIM  16
#define HID       128
#define HEADS     4
#define LAYERS    4
#define NUM_GRAPHS 256
#define HC        (HEADS * HID)   // 512

typedef __attribute__((ext_vector_type(16))) _Float16 v16h;
typedef __attribute__((ext_vector_type(8)))  _Float16 v8h;
typedef __attribute__((ext_vector_type(8)))  float    v8f;

// ---------------------------------------------------------------------------
// Weight prep: W[K,N] f32 row-major  ->  Wt[N,K] f16 (transposed) so the
// WMMA B-operand loads become two contiguous global_load_b128 per lane.
// ---------------------------------------------------------------------------
__global__ void transpose_cast_kernel(const float* __restrict__ in,
                                      _Float16* __restrict__ out, int K, int N)
{
    int i = blockIdx.x * blockDim.x + threadIdx.x;   // i = n*K + k
    if (i >= K * N) return;
    int n = i / K, k = i - n * K;
    out[i] = (_Float16)in[(size_t)k * N + n];
}

// ---------------------------------------------------------------------------
// WMMA GEMM: C[M,N] = act(A[M,K] * B[K,N] + bias), A f32 row-major,
// B pre-transposed f16 [N,K]. K is compile-time (64 or 128, K%32==0).
// 8 waves / block, one 16x16 tile per wave. grid = (M/16, (N/16)/8).
// ACT: 0 = none, 1 = ELU.
// ---------------------------------------------------------------------------
template<int K, int ACT>
__global__ __launch_bounds__(256)
void wmma_gemm_kernel(const float* __restrict__ A, int lda,
                      const _Float16* __restrict__ Bt,
                      float* __restrict__ C, int ldc,
                      const float* __restrict__ bias)
{
    const int lane = threadIdx.x & 31;
    const int wid  = threadIdx.x >> 5;
    const int row0 = blockIdx.x * 16;
    const int col0 = (blockIdx.y * 8 + wid) * 16;

    // Per-lane base pointers (A: M=lane&15 row; B: N=lane&15 column)
    const float*    pa = A  + (size_t)(row0 + (lane & 15)) * lda;
    const _Float16* pb = Bt + (size_t)(col0 + (lane & 15)) * K + ((lane >> 4) << 3);
    const int kaoff = (lane >> 4) << 3;   // +0 / +8

    v8f acc = {};
#pragma unroll
    for (int kk = 0; kk < K; kk += 32) {
        // A 16x32 f16 layout: elems 0..7 = K kb..kb+7, elems 8..15 = kb+16..kb+23
        float af[16];
        const int kb = kk + kaoff;
        *(float4*)(af + 0)  = *(const float4*)(pa + kb);
        *(float4*)(af + 4)  = *(const float4*)(pa + kb + 4);
        *(float4*)(af + 8)  = *(const float4*)(pa + kb + 16);
        *(float4*)(af + 12) = *(const float4*)(pa + kb + 20);
        v16h a;
#pragma unroll
        for (int i = 0; i < 16; ++i) a[i] = (_Float16)af[i];

        // B 32x16: same K layout as A (elems 0..7 = kb..kb+7, 8..15 = kb+16..kb+23)
        // for column N=lane&15 in the transposed f16 buffer.
        v8h b0 = *(const v8h*)(pb + kk);
        v8h b1 = *(const v8h*)(pb + kk + 16);
        v16h b = __builtin_shufflevector(b0, b1, 0, 1, 2, 3, 4, 5, 6, 7,
                                                 8, 9, 10, 11, 12, 13, 14, 15);

        acc = __builtin_amdgcn_wmma_f32_16x16x32_f16(
            false, a, false, b, (short)0, acc, false, false);
    }

    // C/D layout: lanes 0-15 -> (M=row0+r, N=lane); lanes 16-31 -> M=row0+8+r.
    const int n  = col0 + (lane & 15);
    const int mb = row0 + ((lane >> 4) << 3);
    const float bv = bias ? bias[n] : 0.f;
#pragma unroll
    for (int r = 0; r < 8; ++r) {
        float v = acc[r] + bv;
        if (ACT == 1) v = (v > 0.f) ? v : (__expf(v) - 1.f);   // ELU
        C[(size_t)(mb + r) * ldc + n] = v;
    }
}

// ---------------------------------------------------------------------------
__device__ inline void atomicMaxFloat(float* addr, float val)
{
    int* ai = (int*)addr;
    int old = *ai;
    while (__int_as_float(old) < val) {
        int assumed = old;
        old = atomicCAS(ai, assumed, __float_as_int(val));
        if (old == assumed) break;
    }
}

__global__ void fill_kernel(float* p, float v, int n)
{
    int i = blockIdx.x * blockDim.x + threadIdx.x;
    if (i < n) p[i] = v;
}

// ---------------------------------------------------------------------------
// Edge pass 1: logits (e = edge_attr@We recomputed on the fly from LDS),
// leaky-relu, dot with att, atomic segment max. One wave per edge.
// ---------------------------------------------------------------------------
__global__ __launch_bounds__(256)
void edge_logits_kernel(const int* __restrict__ src, const int* __restrict__ dst,
                        const float* __restrict__ edge_attr,
                        const float* __restrict__ We,  // [16,512] this layer
                        const float* __restrict__ att, // [512] this layer
                        const float* __restrict__ xl, const float* __restrict__ xr,
                        float* __restrict__ logits, float* __restrict__ lmax)
{
    __shared__ float sWe[EDGE_DIM * HC];
    __shared__ float sAtt[HC];
    for (int i = threadIdx.x; i < EDGE_DIM * HC; i += blockDim.x) sWe[i] = We[i];
    for (int i = threadIdx.x; i < HC; i += blockDim.x)            sAtt[i] = att[i];
    __syncthreads();

    const int e = blockIdx.x * (blockDim.x >> 5) + (threadIdx.x >> 5);
    if (e >= N_EDGES) return;
    const int lane = threadIdx.x & 31;
    const int s = src[e], d = dst[e];

    float ea[EDGE_DIM];
#pragma unroll
    for (int t = 0; t < EDGE_DIM; ++t) ea[t] = edge_attr[(size_t)e * EDGE_DIM + t];

    const float* pl = xl + (size_t)s * HC;
    const float* pr = xr + (size_t)d * HC;
    const int hc0 = lane * 16;

    float acc = 0.f;
#pragma unroll
    for (int j = 0; j < 16; ++j) {
        const int hc = hc0 + j;
        float ev = 0.f;
#pragma unroll
        for (int t = 0; t < EDGE_DIM; ++t) ev += ea[t] * sWe[t * HC + hc];
        float m = pl[hc] + pr[hc] + ev;
        m = (m > 0.f) ? m : 0.2f * m;          // leaky_relu
        acc += m * sAtt[hc];
    }
    acc += __shfl_xor(acc, 1);
    acc += __shfl_xor(acc, 2);
    acc += __shfl_xor(acc, 4);
    if ((lane & 7) == 0) {
        const int h = lane >> 3;
        logits[(size_t)e * HEADS + h] = acc;
        atomicMaxFloat(&lmax[(size_t)d * HEADS + h], acc);
    }
}

// Edge pass 2: p = exp(logit - lmax[dst]); denom[dst] += p.
__global__ void edge_softmax_kernel(const int* __restrict__ dst,
                                    const float* __restrict__ logits,
                                    const float* __restrict__ lmax,
                                    float* __restrict__ pbuf,
                                    float* __restrict__ denom)
{
    const int i = blockIdx.x * blockDim.x + threadIdx.x;
    if (i >= N_EDGES * HEADS) return;
    const int e = i >> 2, h = i & 3;
    const int d = dst[e];
    const float p = __expf(logits[i] - lmax[(size_t)d * HEADS + h]);
    pbuf[i] = p;
    atomicAdd(&denom[(size_t)d * HEADS + h], p);
}

// Edge pass 3: gat_out[dst] += alpha * xl[src]. One wave per edge.
__global__ __launch_bounds__(256)
void edge_scatter_kernel(const int* __restrict__ src, const int* __restrict__ dst,
                         const float* __restrict__ pbuf, const float* __restrict__ denom,
                         const float* __restrict__ xl, float* __restrict__ go)
{
    const int e = blockIdx.x * (blockDim.x >> 5) + (threadIdx.x >> 5);
    if (e >= N_EDGES) return;
    const int lane = threadIdx.x & 31;
    const int s = src[e], d = dst[e];
    const int h = lane >> 3;
    const float a = pbuf[(size_t)e * HEADS + h] /
                    (denom[(size_t)d * HEADS + h] + 1e-16f);
    const float* pl = xl + (size_t)s * HC;
    float* pg = go + (size_t)d * HC;
    const int hc0 = lane * 16;
#pragma unroll
    for (int j = 0; j < 16; ++j)
        atomicAdd(&pg[hc0 + j], a * pl[hc0 + j]);
}

// ---------------------------------------------------------------------------
// Per-node epilogue: head mean + bias -> LayerNorm -> ELU (+ residual).
// ---------------------------------------------------------------------------
__global__ __launch_bounds__(HID)
void node_finish_kernel(const float* __restrict__ go,
                        const float* __restrict__ b_gat,
                        const float* __restrict__ gamma,
                        const float* __restrict__ beta,
                        const float* __restrict__ res,  // nullable
                        float* __restrict__ hout)
{
    const int n = blockIdx.x, c = threadIdx.x;
    const float* g = go + (size_t)n * HC;
    float v = 0.25f * (g[c] + g[HID + c] + g[2 * HID + c] + g[3 * HID + c]) + b_gat[c];

    __shared__ float red[8];
    float s = v, q = v * v;
    for (int off = 16; off; off >>= 1) {
        s += __shfl_down(s, off);
        q += __shfl_down(q, off);
    }
    const int wid = threadIdx.x >> 5;
    if ((threadIdx.x & 31) == 0) { red[wid] = s; red[4 + wid] = q; }
    __syncthreads();
    if (threadIdx.x == 0) {
        red[0] = red[0] + red[1] + red[2] + red[3];
        red[4] = red[4] + red[5] + red[6] + red[7];
    }
    __syncthreads();
    const float mu  = red[0] * (1.f / HID);
    const float var = red[4] * (1.f / HID) - mu * mu;
    float y = gamma[c] * (v - mu) * rsqrtf(var + 1e-5f) + beta[c];
    y = (y > 0.f) ? y : (__expf(y) - 1.f);   // ELU
    if (res) y += res[(size_t)n * HID + c];
    hout[(size_t)n * HID + c] = y;
}

// ---------------------------------------------------------------------------
// Pooling
// ---------------------------------------------------------------------------
__global__ __launch_bounds__(HID)
void pool_kernel(const float* __restrict__ h, const int* __restrict__ batch,
                 float* __restrict__ sums, float* __restrict__ counts)
{
    const int n = blockIdx.x, c = threadIdx.x;
    const int g = batch[n];
    atomicAdd(&sums[(size_t)g * HID + c], h[(size_t)n * HID + c]);
    if (c == 0) atomicAdd(&counts[g], 1.f);
}

__global__ void pool_div_kernel(const float* __restrict__ sums,
                                const float* __restrict__ counts,
                                float* __restrict__ hg)
{
    const int i = blockIdx.x * blockDim.x + threadIdx.x;
    if (i >= NUM_GRAPHS * HID) return;
    float cnt = counts[i >> 7];
    cnt = (cnt > 1.f) ? cnt : 1.f;
    hg[i] = sums[i] / cnt;
}

// ---------------------------------------------------------------------------
// Readout MLP: one block (128 threads) per graph.
// ---------------------------------------------------------------------------
__global__ __launch_bounds__(HID)
void mlp_kernel(const float* __restrict__ hg,
                const float* __restrict__ W1, const float* __restrict__ b1,
                const float* __restrict__ W2, const float* __restrict__ b2,
                const float* __restrict__ W3, const float* __restrict__ b3,
                float* __restrict__ out)
{
    const int g = blockIdx.x, c = threadIdx.x;
    __shared__ float s0[HID], s1[HID], s2[HID / 2], red[4];

    s0[c] = hg[(size_t)g * HID + c];
    __syncthreads();

    float a1 = b1[c];
    for (int k = 0; k < HID; ++k) a1 += s0[k] * W1[k * HID + c];
    s1[c] = (a1 > 0.f) ? a1 : 0.f;
    __syncthreads();

    if (c < HID / 2) {
        float a2 = b2[c];
        for (int k = 0; k < HID; ++k) a2 += s1[k] * W2[k * (HID / 2) + c];
        s2[c] = (a2 > 0.f) ? a2 : 0.f;
    }
    __syncthreads();

    float t = (c < HID / 2) ? s2[c] * W3[c] : 0.f;
    for (int off = 16; off; off >>= 1) t += __shfl_down(t, off);
    if ((c & 31) == 0) red[c >> 5] = t;
    __syncthreads();
    if (c == 0) out[g] = red[0] + red[1] + red[2] + red[3] + b3[0];
}

// ---------------------------------------------------------------------------
extern "C" void kernel_launch(void* const* d_in, const int* in_sizes, int n_in,
                              void* d_out, int out_size, void* d_ws, size_t ws_size,
                              hipStream_t stream)
{
    (void)in_sizes; (void)n_in; (void)out_size; (void)ws_size;

    const float* x         = (const float*)d_in[0];
    const int*   eidx      = (const int*)  d_in[1];
    const float* edge_attr = (const float*)d_in[2];
    const int*   batch     = (const int*)  d_in[3];
    const float* Win       = (const float*)d_in[4];
    const float* b_in      = (const float*)d_in[5];
    const float* Wl        = (const float*)d_in[6];
    const float* bl        = (const float*)d_in[7];
    const float* Wr        = (const float*)d_in[8];
    const float* br        = (const float*)d_in[9];
    const float* We        = (const float*)d_in[10];
    const float* att       = (const float*)d_in[11];
    const float* b_gat     = (const float*)d_in[12];
    const float* gamma     = (const float*)d_in[13];
    const float* beta      = (const float*)d_in[14];
    const float* Wres      = (const float*)d_in[15];
    const float* bres      = (const float*)d_in[16];
    const float* W1        = (const float*)d_in[17];
    const float* b1        = (const float*)d_in[18];
    const float* W2        = (const float*)d_in[19];
    const float* b2        = (const float*)d_in[20];
    const float* W3        = (const float*)d_in[21];
    const float* b3        = (const float*)d_in[22];

    const int* src = eidx;
    const int* dst = eidx + N_EDGES;

    // Workspace layout
    char* w = (char*)d_ws;
    auto carve = [&](size_t bytes) { char* p = w; w += (bytes + 255) & ~size_t(255); return p; };
    float* h0     = (float*)carve((size_t)N_NODES * HID * 4);
    float* h1     = (float*)carve((size_t)N_NODES * HID * 4);
    float* xl     = (float*)carve((size_t)N_NODES * HC * 4);
    float* xr     = (float*)carve((size_t)N_NODES * HC * 4);
    float* go     = (float*)carve((size_t)N_NODES * HC * 4);
    float* logits = (float*)carve((size_t)N_EDGES * HEADS * 4);
    float* pbuf   = (float*)carve((size_t)N_EDGES * HEADS * 4);
    float* lmax   = (float*)carve((size_t)N_NODES * HEADS * 4);
    float* denom  = (float*)carve((size_t)N_NODES * HEADS * 4);
    float* res    = (float*)carve((size_t)N_NODES * HID * 4);
    float* sums   = (float*)carve((size_t)NUM_GRAPHS * HID * 4);
    float* counts = (float*)carve((size_t)NUM_GRAPHS * 4);
    float* hg     = (float*)carve((size_t)NUM_GRAPHS * HID * 4);
    _Float16* WinT  = (_Float16*)carve((size_t)IN_CH * HID * 2);
    _Float16* WlT   = (_Float16*)carve((size_t)LAYERS * HID * HC * 2);
    _Float16* WrT   = (_Float16*)carve((size_t)LAYERS * HID * HC * 2);
    _Float16* WresT = (_Float16*)carve((size_t)(LAYERS - 1) * HID * HID * 2);

    // ---- weight prep: f32 [K,N] -> f16 [N,K] ----
    {
        int n1 = IN_CH * HID;
        transpose_cast_kernel<<<(n1 + 255) / 256, 256, 0, stream>>>(Win, WinT, IN_CH, HID);
        for (int i = 0; i < LAYERS; ++i) {
            int n2 = HID * HC;
            transpose_cast_kernel<<<(n2 + 255) / 256, 256, 0, stream>>>(
                Wl + (size_t)i * n2, WlT + (size_t)i * n2, HID, HC);
            transpose_cast_kernel<<<(n2 + 255) / 256, 256, 0, stream>>>(
                Wr + (size_t)i * n2, WrT + (size_t)i * n2, HID, HC);
        }
        for (int i = 0; i < LAYERS - 1; ++i) {
            int n3 = HID * HID;
            transpose_cast_kernel<<<(n3 + 255) / 256, 256, 0, stream>>>(
                Wres + (size_t)i * n3, WresT + (size_t)i * n3, HID, HID);
        }
    }

    const int MT = N_NODES / 16;                 // 625 row tiles
    const int edgeBlocks = N_EDGES / 8;

    // Input projection: h0 = elu(x @ Win + b_in)
    wmma_gemm_kernel<IN_CH, 1><<<dim3(MT, (HID / 16) / 8), 256, 0, stream>>>(
        x, IN_CH, WinT, h0, HID, b_in);

    float* hcur = h0;
    float* hnext = h1;

    for (int i = 0; i < LAYERS; ++i) {
        wmma_gemm_kernel<HID, 0><<<dim3(MT, (HC / 16) / 8), 256, 0, stream>>>(
            hcur, HID, WlT + (size_t)i * HID * HC, xl, HC, bl + (size_t)i * HC);
        wmma_gemm_kernel<HID, 0><<<dim3(MT, (HC / 16) / 8), 256, 0, stream>>>(
            hcur, HID, WrT + (size_t)i * HID * HC, xr, HC, br + (size_t)i * HC);

        hipMemsetAsync(go,    0, (size_t)N_NODES * HC * 4, stream);
        hipMemsetAsync(denom, 0, (size_t)N_NODES * HEADS * 4, stream);
        fill_kernel<<<(N_NODES * HEADS + 255) / 256, 256, 0, stream>>>(
            lmax, -1e30f, N_NODES * HEADS);

        edge_logits_kernel<<<edgeBlocks, 256, 0, stream>>>(
            src, dst, edge_attr,
            We + (size_t)i * EDGE_DIM * HC, att + (size_t)i * HC,
            xl, xr, logits, lmax);

        edge_softmax_kernel<<<(N_EDGES * HEADS + 255) / 256, 256, 0, stream>>>(
            dst, logits, lmax, pbuf, denom);

        edge_scatter_kernel<<<edgeBlocks, 256, 0, stream>>>(
            src, dst, pbuf, denom, xl, go);

        if (i > 0) {
            wmma_gemm_kernel<HID, 0><<<dim3(MT, (HID / 16) / 8), 256, 0, stream>>>(
                hcur, HID, WresT + (size_t)(i - 1) * HID * HID, res, HID,
                bres + (size_t)(i - 1) * HID);
        }

        node_finish_kernel<<<N_NODES, HID, 0, stream>>>(
            go, b_gat + (size_t)i * HID, gamma + (size_t)i * HID,
            beta + (size_t)i * HID, (i > 0) ? res : (const float*)nullptr, hnext);

        float* t = hcur; hcur = hnext; hnext = t;
    }

    hipMemsetAsync(sums,   0, (size_t)NUM_GRAPHS * HID * 4, stream);
    hipMemsetAsync(counts, 0, (size_t)NUM_GRAPHS * 4, stream);
    pool_kernel<<<N_NODES, HID, 0, stream>>>(hcur, batch, sums, counts);
    pool_div_kernel<<<(NUM_GRAPHS * HID + 255) / 256, 256, 0, stream>>>(sums, counts, hg);

    mlp_kernel<<<NUM_GRAPHS, HID, 0, stream>>>(
        hg, W1, b1, W2, b2, W3, b3, (float*)d_out);
}
